// MHAInferenceKernel_58643483459739
// MI455X (gfx1250) — compile-verified, matrix operand order checked by reading, not run on hardware
//
#include <hip/hip_runtime.h>
#include <stdint.h>

// The Python harness owns main(), I/O, device malloc/memcpy, and timing.

typedef __attribute__((ext_vector_type(16))) _Float16 v16h;
typedef __attribute__((ext_vector_type(8)))  float    v8f;
typedef __attribute__((ext_vector_type(4)))  int      v4i_t;

union Frag16 { uint4 u[2]; v16h h; };
union HalfBits { _Float16 f; unsigned short u; };

#define B_    4
#define QL_   128
#define H_    32
#define D_    128
#define KV_   8192
#define NS_   4
#define KVS_  (KV_/NS_)     // 2048 keys per split
#define CH_   64            // keys per staged chunk
#define NCH_  (KVS_/CH_)    // 32 chunks
#define HD_   (H_*D_)       // row stride in K/V (4096 halfs)

#define KPAD_ 136           // kbuf row stride (halfs): 16B-aligned, bank-spread
#define VPAD_ 72            // vtbuf/pbuf row stride (halfs): 16B-aligned, bank-spread

// CDNA5 async global->LDS DMA (ASYNCcnt-tracked, no VGPR round trip).
// Probe-learned signature: (global v4i* src, shared v4i* dst, int offset, int cpol)
#if defined(__gfx1250__) && __has_builtin(__builtin_amdgcn_global_load_async_to_lds_b128)
#define USE_ASYNC_LDS 1
typedef __attribute__((address_space(1))) v4i_t gas_v4i;   // int4 in global AS
typedef __attribute__((address_space(3))) v4i_t las_v4i;   // int4 in LDS AS
#else
#define USE_ASYNC_LDS 0
#endif

__device__ __forceinline__ float warp16_max(float v) {
  v = fmaxf(v, __shfl_xor(v, 1, 32));
  v = fmaxf(v, __shfl_xor(v, 2, 32));
  v = fmaxf(v, __shfl_xor(v, 4, 32));
  v = fmaxf(v, __shfl_xor(v, 8, 32));
  return v;
}

// One workgroup = one (b, h, split). 8 waves x 16 q-rows = all 128 q rows.
// K/V chunks staged once into LDS, consumed by all 8 waves. K is double-
// buffered and copied by the async DMA engine one chunk ahead.
__global__ __launch_bounds__(256, 1)
void fa_split_partial(const _Float16* __restrict__ Q,   // [B,QL,H,D]
                      const _Float16* __restrict__ K,   // [B,KV,H,D]
                      const _Float16* __restrict__ V,   // [B,KV,H,D]
                      _Float16* __restrict__ opart,     // [B,H,NS,QL,D] ws
                      float* __restrict__ lse2)         // [B,H,NS,QL]  ws (log2 domain)
{
#if USE_ASYNC_LDS
  __shared__ unsigned short kbuf[2][CH_ * KPAD_];    // K chunks, double-buffered (34 KB)
#else
  __shared__ unsigned short kbuf[1][CH_ * KPAD_];    // K chunk row-major        (17 KB)
#endif
  __shared__ unsigned short vtbuf[D_ * VPAD_];       // V chunk transposed       (18 KB)
  __shared__ unsigned short pbuf[8 * 16 * VPAD_];    // per-wave P staging       (18 KB)

  const int bid = blockIdx.x;
  const int sp  = bid & (NS_ - 1);
  const int h   = (bid >> 2) & (H_ - 1);
  const int b   = bid >> 7;

  const int tid  = threadIdx.x;
  const int wave = tid >> 5;
  const int lane = tid & 31;
  const int g    = lane >> 4;   // half-wave group
  const int n    = lane & 15;   // row (A) / column (B) index within a 16-tile
  const int q0   = wave * 16;

  const float KSC = 0.08838834764831845f * 1.4426950408889634f; // 1/sqrt(D) * log2(e)

  // ---- Q fragments (A-matrix 16x32 f16 layout), registers for whole loop ----
  v16h qa[4];
  {
    const _Float16* qrow = Q + (((size_t)b * QL_ + q0 + n) * H_ + h) * D_;
    #pragma unroll
    for (int ci = 0; ci < 4; ++ci) {
      Frag16 f;
      f.u[0] = *(const uint4*)(qrow + ci * 32 + 8 * g);
      f.u[1] = *(const uint4*)(qrow + ci * 32 + 8 * g + 16);
      qa[ci] = f.h;
    }
  }

  // all-ones f16 B fragment: one WMMA against it computes row sums of P,
  // replicated into every lane of the C tile (replaces ds_bpermute reductions)
  v16h onesv;
  {
    Frag16 of;
    uint4 ov; ov.x = ov.y = ov.z = ov.w = 0x3C003C00u;  // (1.0h, 1.0h)
    of.u[0] = ov; of.u[1] = ov;
    onesv = of.h;
  }

  v8f zero8 = {0.f, 0.f, 0.f, 0.f, 0.f, 0.f, 0.f, 0.f};
  v8f acc[8];
  #pragma unroll
  for (int dt = 0; dt < 8; ++dt) acc[dt] = zero8;
  float mrow[8], lrow[8];
  #pragma unroll
  for (int i = 0; i < 8; ++i) { mrow[i] = -1e30f; lrow[i] = 0.f; }

  const size_t kvbase = ((size_t)b * KV_) * HD_ + (size_t)h * D_;
  // per-thread K staging geometry: 4 units of (row, octet)
  const int kr   = tid >> 4;          // rows kr, kr+16, kr+32, kr+48
  const int koff = (tid & 15) * 8;

#if USE_ASYNC_LDS
  // prologue: async-copy chunk 0 of K into kbuf[0]
  {
    const int kv0 = sp * KVS_;
    #pragma unroll
    for (int j = 0; j < 4; ++j) {
      const int r = kr + 16 * j;
      __builtin_amdgcn_global_load_async_to_lds_b128(
          (gas_v4i*)(K + kvbase + (size_t)(kv0 + r) * HD_ + koff),
          (las_v4i*)&kbuf[0][r * KPAD_ + koff], 0, 0);
    }
  }
#endif

  for (int ch = 0; ch < NCH_; ++ch) {
    const int kv0 = sp * KVS_ + ch * CH_;
    const int p   = ch & 1;
    __syncthreads();  // previous chunk fully consumed (vtbuf/pbuf reusable)

#if !USE_ASYNC_LDS
    // ---- stage K chunk (register bounce fallback) ----
    #pragma unroll
    for (int j = 0; j < 4; ++j) {
      const int r = kr + 16 * j;
      const size_t gsrc = kvbase + (size_t)(kv0 + r) * HD_ + koff;
      uint4 kd = *(const uint4*)(K + gsrc);
      *(uint4*)(&kbuf[0][r * KPAD_ + koff]) = kd;
      if (ch + 1 < NCH_)
        __builtin_prefetch(K + gsrc + (size_t)CH_ * HD_, 0, 1);
    }
#endif
    // ---- stage V chunk transposed, packed column-pair b32 stores ----
    #pragma unroll
    for (int j = 0; j < 2; ++j) {
      const int e   = tid + 256 * j;           // 0..511 -> 32 row-pairs x 16 octets
      const int rp  = e & 31;                   // kv row pair -> rows 2rp, 2rp+1
      const int off = (e >> 5) * 8;             // d octet
      const _Float16* vsrc = V + kvbase + (size_t)(kv0 + 2 * rp) * HD_ + off;
      uint4 v0 = *(const uint4*)(vsrc);
      uint4 v1 = *(const uint4*)(vsrc + HD_);
      const unsigned int* va = (const unsigned int*)&v0;
      const unsigned int* vb = (const unsigned int*)&v1;
      #pragma unroll
      for (int jj = 0; jj < 4; ++jj) {
        unsigned int x = va[jj], y = vb[jj];
        unsigned int lo = (x & 0x0000ffffu) | (y << 16);        // d=off+2jj
        unsigned int hi = (x >> 16) | (y & 0xffff0000u);        // d=off+2jj+1
        *(unsigned int*)(&vtbuf[(off + 2 * jj)     * VPAD_ + 2 * rp]) = lo;
        *(unsigned int*)(&vtbuf[(off + 2 * jj + 1) * VPAD_ + 2 * rp]) = hi;
      }
      if (ch + 1 < NCH_)
        __builtin_prefetch(vsrc + (size_t)CH_ * HD_, 0, 1);
    }
#if USE_ASYNC_LDS
    // drain this wave's outstanding async K copies (chunk ch), then rendezvous
    asm volatile("s_wait_asynccnt 0x0" ::: "memory");
#endif
    __syncthreads();

#if USE_ASYNC_LDS
    // kick off async K copy for chunk ch+1 into the other buffer; the DMA
    // overlaps with this chunk's WMMA work and next chunk's V staging
    if (ch + 1 < NCH_) {
      #pragma unroll
      for (int j = 0; j < 4; ++j) {
        const int r = kr + 16 * j;
        __builtin_amdgcn_global_load_async_to_lds_b128(
            (gas_v4i*)(K + kvbase + (size_t)(kv0 + CH_ + r) * HD_ + koff),
            (las_v4i*)&kbuf[p ^ 1][r * KPAD_ + koff], 0, 0);
      }
    }
    const unsigned short* kcur = kbuf[p];
#else
    const unsigned short* kcur = kbuf[0];
#endif

    // ---- scores: S = Q * K^T : 16 x 64 in four 16x16 tiles ----
    v8f sc[4];
    #pragma unroll
    for (int t = 0; t < 4; ++t) sc[t] = zero8;
    #pragma unroll
    for (int ci = 0; ci < 4; ++ci) {
      const int dd = ci * 32 + 8 * g;
      #pragma unroll
      for (int t = 0; t < 4; ++t) {
        Frag16 kb;
        kb.u[0] = *(const uint4*)(&kcur[(16 * t + n) * KPAD_ + dd]);
        kb.u[1] = *(const uint4*)(&kcur[(16 * t + n) * KPAD_ + dd + 16]);
        sc[t] = __builtin_amdgcn_wmma_f32_16x16x32_f16(false, qa[ci], false, kb.h,
                                                       (short)0, sc[t], false, false);
      }
    }

    // ---- online softmax over 64 new keys (base-2); only max needs shuffles ----
    unsigned short* pw = &pbuf[wave * 16 * VPAD_];
    #pragma unroll
    for (int i = 0; i < 8; ++i) {            // C-layout: element i = q row 8*g + i
      float cm = fmaxf(fmaxf(sc[0][i], sc[1][i]), fmaxf(sc[2][i], sc[3][i]));
      cm = warp16_max(cm) * KSC;
      float mn = fmaxf(mrow[i], cm);
      float alpha = __builtin_amdgcn_exp2f(mrow[i] - mn);
      mrow[i] = mn;
      lrow[i] *= alpha;                      // row-sum of new keys added after WMMA
      #pragma unroll
      for (int dt = 0; dt < 8; ++dt) acc[dt][i] *= alpha;
      #pragma unroll
      for (int t = 0; t < 4; ++t) {
        float pv = __builtin_amdgcn_exp2f(sc[t][i] * KSC - mn);
        HalfBits c; c.f = (_Float16)pv;
        pw[(8 * g + i) * VPAD_ + 16 * t + n] = c.u;
      }
    }
    // same-wave LDS RAW: DS ops are in-order per wave; drain the counter
    asm volatile("s_wait_dscnt 0x0" ::: "memory");

    // ---- P fragments (A 16x32 layout), kv halves 0..31 and 32..63 ----
    Frag16 pf0, pf1;
    {
      const unsigned short* pr = &pbuf[wave * 16 * VPAD_ + n * VPAD_];
      pf0.u[0] = *(const uint4*)(pr + 8 * g);
      pf0.u[1] = *(const uint4*)(pr + 16 + 8 * g);
      pf1.u[0] = *(const uint4*)(pr + 32 + 8 * g);
      pf1.u[1] = *(const uint4*)(pr + 48 + 8 * g);
    }

    // row sums of P via WMMA against all-ones B (replicated across columns)
    v8f lsum = zero8;
    lsum = __builtin_amdgcn_wmma_f32_16x16x32_f16(false, pf0.h, false, onesv,
                                                  (short)0, lsum, false, false);
    lsum = __builtin_amdgcn_wmma_f32_16x16x32_f16(false, pf1.h, false, onesv,
                                                  (short)0, lsum, false, false);
    #pragma unroll
    for (int i = 0; i < 8; ++i) lrow[i] += lsum[i];

    // ---- O += P * V ----
    #pragma unroll
    for (int dt = 0; dt < 8; ++dt) {
      const unsigned short* vr = &vtbuf[(dt * 16 + n) * VPAD_];
      Frag16 vf0, vf1;
      vf0.u[0] = *(const uint4*)(vr + 8 * g);
      vf0.u[1] = *(const uint4*)(vr + 16 + 8 * g);
      vf1.u[0] = *(const uint4*)(vr + 32 + 8 * g);
      vf1.u[1] = *(const uint4*)(vr + 48 + 8 * g);
      acc[dt] = __builtin_amdgcn_wmma_f32_16x16x32_f16(false, pf0.h, false, vf0.h,
                                                       (short)0, acc[dt], false, false);
      acc[dt] = __builtin_amdgcn_wmma_f32_16x16x32_f16(false, pf1.h, false, vf1.h,
                                                       (short)0, acc[dt], false, false);
    }
  }

  // ---- epilogue: normalize, write per-split partial output + log2-LSE ----
  const size_t obase = ((((size_t)b * H_ + h) * NS_ + sp) * QL_ + q0) * D_;
  #pragma unroll
  for (int i = 0; i < 8; ++i) {
    float rl = __builtin_amdgcn_rcpf(lrow[i]);
    #pragma unroll
    for (int dt = 0; dt < 8; ++dt) {
      opart[obase + (size_t)(8 * g + i) * D_ + dt * 16 + n] =
          (_Float16)(acc[dt][i] * rl);
    }
  }
  const size_t lbase = (((size_t)b * H_ + h) * NS_ + sp) * QL_ + q0;
  #pragma unroll
  for (int i = 0; i < 8; ++i) {
    if (n == i) lse2[lbase + 8 * g + i] =
        __builtin_amdgcn_logf(lrow[i]) + mrow[i];   // log2(l) + m_hat
  }
}

// Combine the 4 splits: softmax over split LSEs (base-2 == base-e softmax).
__global__ __launch_bounds__(128, 4)
void fa_combine(const _Float16* __restrict__ opart,  // [B,H,NS,QL,D]
                const float* __restrict__ lse2,      // [B,H,NS,QL]
                _Float16* __restrict__ out)          // [B,QL,H,D]
{
  const int row = blockIdx.x;              // over B*QL*H
  const int h = row % H_;
  const int q = (row / H_) % QL_;
  const int b = row / (H_ * QL_);
  const int d = threadIdx.x;

  const size_t lb = (((size_t)b * H_ + h) * NS_) * QL_ + q;
  float ls0 = lse2[lb];
  float ls1 = lse2[lb + QL_];
  float ls2 = lse2[lb + 2 * QL_];
  float ls3 = lse2[lb + 3 * QL_];
  float mx = fmaxf(fmaxf(ls0, ls1), fmaxf(ls2, ls3));
  float w0 = __builtin_amdgcn_exp2f(ls0 - mx);
  float w1 = __builtin_amdgcn_exp2f(ls1 - mx);
  float w2 = __builtin_amdgcn_exp2f(ls2 - mx);
  float w3 = __builtin_amdgcn_exp2f(ls3 - mx);
  float ws = w0 + w1 + w2 + w3;

  const size_t pb = ((((size_t)b * H_ + h) * NS_) * QL_ + q) * (size_t)D_ + d;
  const size_t sstr = (size_t)QL_ * D_;
  float o = w0 * (float)opart[pb]
          + w1 * (float)opart[pb + sstr]
          + w2 * (float)opart[pb + 2 * sstr]
          + w3 * (float)opart[pb + 3 * sstr];
  out[(((size_t)b * QL_ + q) * H_ + h) * D_ + d] = (_Float16)(o / ws);
}

extern "C" void kernel_launch(void* const* d_in, const int* in_sizes, int n_in,
                              void* d_out, int out_size, void* d_ws, size_t ws_size,
                              hipStream_t stream) {
  const _Float16* Q = (const _Float16*)d_in[0];
  const _Float16* K = (const _Float16*)d_in[1];
  const _Float16* V = (const _Float16*)d_in[2];
  // d_in[3] (glse) and d_in[4] (Output_partial) are scratch in the reference;
  // inputs must not be mutated, so partials live in d_ws instead.
  _Float16* out = (_Float16*)d_out;

  _Float16* opart = (_Float16*)d_ws;                              // 16 MiB
  const size_t opartBytes = (size_t)B_ * H_ * NS_ * QL_ * D_ * sizeof(_Float16);
  float* lse2 = (float*)((char*)d_ws + opartBytes);               // 256 KiB

  fa_split_partial<<<B_ * H_ * NS_, 256, 0, stream>>>(Q, K, V, opart, lse2);
  fa_combine<<<B_ * QL_ * H_, 128, 0, stream>>>(opart, lse2, out);
}
